// FloatingRegionScore_45732811768485
// MI455X (gfx1250) — 8-bit matrix operand order repaired and checked by reading, compiled, not run on hardware
//
#include <hip/hip_runtime.h>

#define NC   19
#define H    512
#define W    1024
#define HWSZ (H * W)
#define RAD  16
#define SEG  64
#define EPSF 1e-6f
#define INV_LOGC (1.0f / 2.9444389791664403f)

typedef __attribute__((ext_vector_type(8))) int v8i;

__device__ __forceinline__ int winsz(int i, int n) {
    int lo = i - RAD; if (lo < 0) lo = 0;
    int hi = i + RAD; if (hi > n - 1) hi = n - 1;
    return hi - lo + 1;
}

__global__ void __launch_bounds__(256)
fr_ent_pred(const float* __restrict__ logit,
            float* __restrict__ ent, unsigned char* __restrict__ pred)
{
    int idx = blockIdx.x * blockDim.x + threadIdx.x;
    if (idx >= HWSZ) return;
    float x[NC];
    float m = -3.0e38f; int am = 0;
#pragma unroll
    for (int c = 0; c < NC; ++c) {
        float v = logit[c * HWSZ + idx];
        x[c] = v;
        if (v > m) { m = v; am = c; }
    }
    float s = 0.f;
#pragma unroll
    for (int c = 0; c < NC; ++c) { float e = __expf(x[c] - m); x[c] = e; s += e; }
    float inv = 1.0f / s;
    float acc = 0.f;
#pragma unroll
    for (int c = 0; c < NC; ++c) { float p = x[c] * inv; acc -= p * __logf(p + EPSF); }
    ent[idx]  = acc * INV_LOGC;
    pred[idx] = (unsigned char)am;
}

__global__ void __launch_bounds__(256)
fr_col_ent(const float* __restrict__ ent, float* __restrict__ colsum)
{
    int idx = blockIdx.x * blockDim.x + threadIdx.x;
    if (idx >= HWSZ) return;
    int h = idx >> 10;
    int w = idx & (W - 1);
    float s = 0.f;
#pragma unroll
    for (int dh = -RAD; dh <= RAD; ++dh) {
        int r = h + dh;
        if (r >= 0 && r < H) s += ent[r * W + w];
    }
    colsum[idx] = s;
}

__global__ void __launch_bounds__(128)
fr_row_ent(const float* __restrict__ colsum, float* __restrict__ runc)
{
    int t = blockIdx.x * blockDim.x + threadIdx.x;
    if (t >= H * (W / SEG)) return;
    int h  = t / (W / SEG);
    int s0 = (t % (W / SEG)) * SEG;
    const float* row = colsum + h * W;
    float sum = 0.f;
    for (int x = s0 - RAD; x <= s0 + RAD; ++x)
        if (x >= 0 && x < W) sum += row[x];
    for (int w = s0; w < s0 + SEG; ++w) {
        runc[h * W + w] = sum;
        int xin = w + RAD + 1, xout = w - RAD;
        if (xin  < W)  sum += row[xin];
        if (xout >= 0) sum -= row[xout];
    }
}

__global__ void __launch_bounds__(32)
fr_col_count(const unsigned char* __restrict__ pred,
             unsigned char* __restrict__ colcnt)
{
    __shared__ unsigned char tilebuf[256];

    int tile = blockIdx.x;
    int th = tile >> 6, tw = tile & 63;
    int r0 = th * 16, w0 = tw * 16;
    int lane = threadIdx.x & 31;
    int hi = (lane >> 4) & 1;
    int mm = lane & 15;

    v8i a;
#pragma unroll
    for (int v = 0; v < 8; ++v) {
        int kbase = ((v & 4) ? 32 : 0) + (hi ? 16 : 0) + (v & 3) * 4;
        unsigned int word = 0;
#pragma unroll
        for (int b = 0; b < 4; ++b) {
            int k = kbase + b;
            if (k >= mm && k <= mm + 32) word |= (1u << (8 * b));
        }
        a[v] = (int)word;
    }

    unsigned int pp[8];
#pragma unroll
    for (int v = 0; v < 8; ++v) {
        int kbase = ((v & 4) ? 32 : 0) + (hi ? 16 : 0) + (v & 3) * 4;
        unsigned int word = 0;
#pragma unroll
        for (int b = 0; b < 4; ++b) {
            int row = r0 - 16 + kbase + b;
            unsigned int pv = 0x7Fu;
            if (row >= 0 && row < H) pv = pred[row * W + w0 + mm];
            word |= pv << (8 * b);
        }
        pp[v] = word;
    }

    v8i cz = {0, 0, 0, 0, 0, 0, 0, 0};
    for (int c = 0; c < NC; ++c) {
        unsigned int cb = 0x01010101u * (unsigned)c;
        v8i bm;
#pragma unroll
        for (int v = 0; v < 8; ++v) {
            unsigned int x  = pp[v] ^ cb;
            unsigned int t  = (x & 0x7F7F7F7Fu) + 0x7F7F7F7Fu;
            unsigned int nz = (t | x) & 0x80808080u;
            bm[v] = (int)((~nz & 0x80808080u) >> 7);
        }
        v8i d = __builtin_amdgcn_wmma_i32_16x16x64_iu8(false, a, false, bm, cz,
                                                       false, false);
        __syncthreads();
#pragma unroll
        for (int g = 0; g < 8; ++g)
            tilebuf[(g + 8 * hi) * 16 + mm] = (unsigned char)d[g];
        __syncthreads();

        unsigned int d0 = ((const unsigned int*)tilebuf)[lane];
        unsigned int d1 = ((const unsigned int*)tilebuf)[lane + 32];
        int colo  = (lane & 3) * 4;
        int row_a = r0 + (lane >> 2);
        int row_b = row_a + 8;
        *(unsigned int*)(colcnt + (size_t)c * HWSZ + row_a * W + w0 + colo) = d0;
        *(unsigned int*)(colcnt + (size_t)c * HWSZ + row_b * W + w0 + colo) = d1;
    }
}

__global__ void __launch_bounds__(32)
fr_final(const unsigned char* __restrict__ colcnt,
         const float* __restrict__ runc,
         float* __restrict__ out)
{
    int tile = blockIdx.x;
    int th = tile >> 6, tw = tile & 63;
    int r0 = th * 16, w0 = tw * 16;
    int lane = threadIdx.x & 31;
    int hi = (lane >> 4) & 1;
    int nn = lane & 15;

    v8i bmat;
#pragma unroll
    for (int v = 0; v < 8; ++v) {
        int kbase = ((v & 4) ? 32 : 0) + ((v & 2) ? 16 : 0) + (hi ? 8 : 0) + (v & 1) * 4;
        unsigned int word = 0;
#pragma unroll
        for (int b = 0; b < 4; ++b) {
            int k = kbase + b;
            if (k >= nn && k <= nn + 32) word |= (1u << (8 * b));
        }
        bmat[v] = (int)word;
    }

    int wpix = w0 + nn;
    float invc[8];
#pragma unroll
    for (int g = 0; g < 8; ++g) {
        int row = r0 + g + hi * 8;
        invc[g] = 1.0f / (float)(winsz(row, H) * winsz(wpix, W));
    }

    v8i cz = {0, 0, 0, 0, 0, 0, 0, 0};
    float imp[8] = {0.f, 0.f, 0.f, 0.f, 0.f, 0.f, 0.f, 0.f};

    for (int c = 0; c < NC; ++c) {
        const unsigned char* base = colcnt + (size_t)c * HWSZ + (r0 + nn) * W;
        if (c + 1 < NC) __builtin_prefetch(base + HWSZ, 0, 1);
        v8i a;
#pragma unroll
        for (int v = 0; v < 8; ++v) {
            int kbase = ((v & 4) ? 32 : 0) + ((v & 2) ? 16 : 0) + (hi ? 8 : 0) + (v & 1) * 4;
            int wc = w0 - 16 + kbase;
            unsigned int word = 0;
            if (wc >= 0 && wc <= W - 4)
                word = *(const unsigned int*)(base + wc);
            a[v] = (int)word;
        }
        v8i d = __builtin_amdgcn_wmma_i32_16x16x64_iu8(false, a, false, bmat, cz,
                                                       false, false);
#pragma unroll
        for (int g = 0; g < 8; ++g) {
            float dist = (float)d[g] * invc[g];
            imp[g] -= dist * __logf(dist + EPSF);
        }
    }

#pragma unroll
    for (int g = 0; g < 8; ++g) {
        int row = r0 + g + hi * 8;
        int o = row * W + wpix;
        float impv = imp[g] * INV_LOGC;
        float punc = runc[o] * invc[g];
        out[o]            = impv * punc;
        out[HWSZ + o]     = impv;
        out[2 * HWSZ + o] = punc;
    }
}

extern "C" void kernel_launch(void* const* d_in, const int* in_sizes, int n_in,
                              void* d_out, int out_size, void* d_ws, size_t ws_size,
                              hipStream_t stream)
{
    const float* logit = (const float*)d_in[0];
    float* out = (float*)d_out;

    char* ws = (char*)d_ws;
    float* ent            = (float*)(ws);
    float* colsum         = (float*)(ws + (size_t)HWSZ * 4);
    float* runc           = (float*)(ws + (size_t)HWSZ * 8);
    unsigned char* pred   = (unsigned char*)(ws + (size_t)HWSZ * 12);
    unsigned char* colcnt = (unsigned char*)(ws + (size_t)HWSZ * 12 + HWSZ);

    const int ntiles = (H / 16) * (W / 16);
    const int nrseg  = H * (W / SEG);

    fr_ent_pred <<<HWSZ / 256, 256, 0, stream>>>(logit, ent, pred);
    fr_col_ent  <<<HWSZ / 256, 256, 0, stream>>>(ent, colsum);
    fr_row_ent  <<<nrseg / 128, 128, 0, stream>>>(colsum, runc);
    fr_col_count<<<ntiles,      32, 0, stream>>>(pred, colcnt);
    fr_final    <<<ntiles,      32, 0, stream>>>(colcnt, runc, out);
}
